// NIUscaled_dot_product_attention_16423954939916
// MI455X (gfx1250) — matrix operand order repaired and checked by reading, not compiled, not run on hardware
//
#include <hip/hip_runtime.h>
#include <hip/hip_bf16.h>

// ---------------------------------------------------------------------------
// Flash attention (causal) for B=2,H=16,S=2048,D=64, fp32 in/out.
// bf16 WMMA (v_wmma_f32_16x16x32_bf16) with fp32 accumulation, online softmax
// reduced with DPP8/permlane16 (no LDS traffic), double-buffered
// GLOBAL_LOAD_ASYNC_TO_LDS staging of K/V (ASYNCcnt pipeline).
// ---------------------------------------------------------------------------

typedef __attribute__((ext_vector_type(16))) __bf16 v16bf;
typedef __attribute__((ext_vector_type(8)))  float  v8f;
typedef int v4i __attribute__((vector_size(16)));   // matches async-LDS builtin

#define S_LEN   2048
#define D_DIM   64
#define QTILE   16                 // queries per wave
#define KTILE   32                 // keys per iteration
#define NWAVES  4
#define BLOCKQ  (QTILE * NWAVES)   // 64 queries per workgroup
#define LOG2E   1.4426950408889634f

#define AS1 __attribute__((address_space(1)))
#define AS3 __attribute__((address_space(3)))

#if __has_builtin(__builtin_amdgcn_global_load_async_to_lds_b128)
#define FA_ASYNC_STAGE 1
#else
#define FA_ASYNC_STAGE 0
#endif

#if __has_builtin(__builtin_amdgcn_mov_dpp8) && __has_builtin(__builtin_amdgcn_permlane16)
#define FA_DPP_RED 1
#else
#define FA_DPP_RED 0
#endif

// ---- cross-lane row reductions over 16-lane groups ------------------------
#if FA_DPP_RED
#define DPP8_SEL(a,b,c,d,e,f,g,h) \
    ((a)|((b)<<3)|((c)<<6)|((d)<<9)|((e)<<12)|((f)<<15)|((g)<<18)|((h)<<21))
#define FXOR1(x) __builtin_bit_cast(float, __builtin_amdgcn_mov_dpp8( \
    __builtin_bit_cast(int, (x)), DPP8_SEL(1,0,3,2,5,4,7,6)))
#define FXOR2(x) __builtin_bit_cast(float, __builtin_amdgcn_mov_dpp8( \
    __builtin_bit_cast(int, (x)), DPP8_SEL(2,3,0,1,6,7,4,5)))
#define FXOR4(x) __builtin_bit_cast(float, __builtin_amdgcn_mov_dpp8( \
    __builtin_bit_cast(int, (x)), DPP8_SEL(4,5,6,7,0,1,2,3)))
__device__ __forceinline__ float fxor8(float x) {      // lane ^ 8 within 16
    int i = __builtin_bit_cast(int, x);
    int r = __builtin_amdgcn_permlane16(i, i, 0xFEDCBA98u, 0x76543210u,
                                        false, false);
    return __builtin_bit_cast(float, r);
}
__device__ __forceinline__ float row_max16(float v) {
    v = fmaxf(v, FXOR1(v));
    v = fmaxf(v, FXOR2(v));
    v = fmaxf(v, FXOR4(v));
    v = fmaxf(v, fxor8(v));
    return v;
}
__device__ __forceinline__ float row_sum16(float v) {
    v += FXOR1(v);
    v += FXOR2(v);
    v += FXOR4(v);
    v += fxor8(v);
    return v;
}
#else
__device__ __forceinline__ float row_max16(float v) {
    v = fmaxf(v, __shfl_xor(v, 1));
    v = fmaxf(v, __shfl_xor(v, 2));
    v = fmaxf(v, __shfl_xor(v, 4));
    v = fmaxf(v, __shfl_xor(v, 8));
    return v;
}
__device__ __forceinline__ float row_sum16(float v) {
    v += __shfl_xor(v, 1);
    v += __shfl_xor(v, 2);
    v += __shfl_xor(v, 4);
    v += __shfl_xor(v, 8);
    return v;
}
#endif

__global__ __launch_bounds__(128) void
fa_fwd_kernel(const float* __restrict__ Q, const float* __restrict__ K,
              const float* __restrict__ V, float* __restrict__ O)
{
    // bf16 operand tiles + per-wave P scratch
    __shared__ __attribute__((aligned(32))) __bf16 Kl [KTILE * D_DIM];    // [key][d]
    __shared__ __attribute__((aligned(32))) __bf16 VTl[D_DIM * KTILE];    // [d][key]
    __shared__ __attribute__((aligned(32))) __bf16 Pl [NWAVES][QTILE * KTILE];
#if FA_ASYNC_STAGE
    // double-buffered raw f32 staging, filled by async global->LDS DMA
    __shared__ __attribute__((aligned(16))) float Kraw[2][KTILE * D_DIM]; // 2 x 8KB
    __shared__ __attribute__((aligned(16))) float Vraw[2][KTILE * D_DIM]; // 2 x 8KB
#endif

    const int tid  = threadIdx.x;
    const int lane = tid & 31;
    const int wave = tid >> 5;
    const int ln16 = lane & 15;
    const int hi   = lane >> 4;            // 0: lanes 0-15, 1: lanes 16-31

    const int bh     = blockIdx.x >> 5;    // 32 q-blocks per (b,h)
    const int qblock = blockIdx.x & 31;
    const int q0wg   = qblock * BLOCKQ;
    const int q0     = q0wg + wave * QTILE;

    const float* Qp = Q + (size_t)bh * S_LEN * D_DIM;
    const float* Kp = K + (size_t)bh * S_LEN * D_DIM;
    const float* Vp = V + (size_t)bh * S_LEN * D_DIM;
    float*       Op = O + (size_t)bh * S_LEN * D_DIM;

    const float scale = 0.125f;            // 1/sqrt(64)

    // per-thread staging coordinates: 16 consecutive floats of the 32x64 tile
    const int sbase = tid * 16;
    const int skrow = sbase >> 6;          // key row within tile
    const int sdcol = sbase & 63;          // d column start

    // ---- Q tile -> two 16x32 bf16 A-fragments, pre-scaled -----------------
    // A fragment in this code's contiguous k order (same as B): lane row = lane%16; element e maps to
    // k = 16*hi + e
    v16bf qa[2];
    {
        const float* qrow = Qp + (size_t)(q0 + ln16) * D_DIM;
        #pragma unroll
        for (int c = 0; c < 2; ++c) {
            #pragma unroll
            for (int half = 0; half < 2; ++half) {
                const int dbase = c * 32 + hi * 16 + half * 8;
                #pragma unroll
                for (int i = 0; i < 8; ++i)
                    qa[c][half * 8 + i] = (__bf16)(qrow[dbase + i] * scale);
            }
        }
    }

    // ---- softmax state + O accumulator (C layout: row = r + 8*hi) --------
    float m[8], l[8];
    v8f acc[4];
    #pragma unroll
    for (int r = 0; r < 8; ++r) { m[r] = -1e30f; l[r] = 0.0f; }
    #pragma unroll
    for (int f = 0; f < 4; ++f) {
        #pragma unroll
        for (int r = 0; r < 8; ++r) acc[f][r] = 0.0f;
    }

    const int ktiles = (q0wg + BLOCKQ) / KTILE;   // causal bound for whole WG

#if FA_ASYNC_STAGE
    // issue async DMA for tile `t` into buffer `b`: 8 x b128 per thread-lane,
    // i.e. ASYNCcnt += 8 per wave
    auto issue_tile = [&](int t, int b) {
        const int k0 = t * KTILE;
        const float* kg = Kp + (size_t)(k0 + skrow) * D_DIM + sdcol;
        const float* vg = Vp + (size_t)(k0 + skrow) * D_DIM + sdcol;
        float* kl = &Kraw[b][sbase];
        float* vl = &Vraw[b][sbase];
        #pragma unroll
        for (int i = 0; i < 4; ++i) {
            __builtin_amdgcn_global_load_async_to_lds_b128(
                (AS1 v4i*)(kg + i * 4), (AS3 v4i*)(kl + i * 4), 0, 0);
            __builtin_amdgcn_global_load_async_to_lds_b128(
                (AS1 v4i*)(vg + i * 4), (AS3 v4i*)(vl + i * 4), 0, 0);
        }
    };
    issue_tile(0, 0);                      // prime the pipeline
#endif

    for (int t = 0; t < ktiles; ++t) {
        const int k0  = t * KTILE;
        const int buf = t & 1;
        (void)buf;

#if FA_ASYNC_STAGE
        // overlap: next tile's HBM traffic in flight during convert+compute
        if (t + 1 < ktiles) {
            issue_tile(t + 1, buf ^ 1);
            asm volatile("s_wait_asynccnt 0x8" ::: "memory");  // tile t landed
        } else {
            asm volatile("s_wait_asynccnt 0x0" ::: "memory");
        }
        __syncthreads();   // all waves done reading previous Kl/VTl
        {
            // wave-local convert: each wave converts exactly what it DMA'd
            const float* kr = &Kraw[buf][sbase];
            #pragma unroll
            for (int i = 0; i < 16; ++i)
                Kl[sbase + i] = (__bf16)kr[i];
            const float* vr = &Vraw[buf][sbase];
            #pragma unroll
            for (int i = 0; i < 16; ++i)
                VTl[(sdcol + i) * KTILE + skrow] = (__bf16)vr[i];  // transpose
        }
        __syncthreads();
#else
        __syncthreads();
        {   // synchronous staging fallback
            const float* kg = Kp + (size_t)(k0 + skrow) * D_DIM + sdcol;
            #pragma unroll
            for (int i = 0; i < 16; ++i)
                Kl[sbase + i] = (__bf16)kg[i];
            const float* vg = Vp + (size_t)(k0 + skrow) * D_DIM + sdcol;
            #pragma unroll
            for (int i = 0; i < 16; ++i)
                VTl[(sdcol + i) * KTILE + skrow] = (__bf16)vg[i];
        }
        __syncthreads();
        if (t + 1 < ktiles) {
            __builtin_prefetch(Kp + (size_t)(k0 + KTILE) * D_DIM + sbase, 0, 1);
            __builtin_prefetch(Vp + (size_t)(k0 + KTILE) * D_DIM + sbase, 0, 1);
        }
#endif

        if (k0 < q0 + QTILE) {   // wave-uniform causal skip (EXEC stays all-1s)
            // ---- S = Q K^T : 2 key sub-tiles x 2 d-chunk WMMAs -----------
            v8f s[2];
            #pragma unroll
            for (int j = 0; j < 2; ++j) {
                v8f c = {};
                #pragma unroll
                for (int cc = 0; cc < 2; ++cc) {
                    // B fragment: lane col = key = 16j + lane%16,
                    // k-dim = 32*cc + 16*hi + e  -> contiguous 16 bf16 in LDS
                    const v16bf* bp = (const v16bf*)
                        &Kl[(j * 16 + ln16) * D_DIM + cc * 32 + hi * 16];
                    c = __builtin_amdgcn_wmma_f32_16x16x32_bf16(
                            false, qa[cc], false, *bp, (short)0, c, false, false);
                }
                s[j] = c;
            }

            // ---- causal mask: only the diagonal-crossing tile needs it ---
            if (k0 + KTILE > q0) {     // wave-uniform
                #pragma unroll
                for (int j = 0; j < 2; ++j) {
                    #pragma unroll
                    for (int r = 0; r < 8; ++r) {
                        const int q = q0 + r + 8 * hi;
                        const int k = k0 + j * 16 + ln16;
                        if (k > q) s[j][r] = -1e30f;
                    }
                }
            }

            // ---- online softmax: DPP8/permlane16 row reductions ----------
            #pragma unroll
            for (int r = 0; r < 8; ++r) {
                const float v  = row_max16(fmaxf(s[0][r], s[1][r]));
                const float mn = fmaxf(m[r], v);
                const float corr = __builtin_amdgcn_exp2f((m[r] - mn) * LOG2E);
                m[r] = mn;
                s[0][r] = __builtin_amdgcn_exp2f((s[0][r] - mn) * LOG2E);
                s[1][r] = __builtin_amdgcn_exp2f((s[1][r] - mn) * LOG2E);
                const float rs = row_sum16(s[0][r] + s[1][r]);
                l[r] = l[r] * corr + rs;
                #pragma unroll
                for (int f = 0; f < 4; ++f) acc[f][r] *= corr;
            }

            // ---- P: C-layout -> A-layout via wave-local LDS transpose ----
            __bf16* pl = &Pl[wave][0];
            #pragma unroll
            for (int j = 0; j < 2; ++j)
                #pragma unroll
                for (int r = 0; r < 8; ++r)
                    pl[(r + 8 * hi) * KTILE + j * 16 + ln16] = (__bf16)s[j][r];
            asm volatile("s_wait_dscnt 0" ::: "memory");   // wave-local RAW on LDS

            v16bf pa;
            #pragma unroll
            for (int half = 0; half < 2; ++half) {
                const int kb = hi * 16 + half * 8;
                #pragma unroll
                for (int i = 0; i < 8; ++i)
                    pa[half * 8 + i] = pl[ln16 * KTILE + kb + i];
            }

            // ---- O += P V : 4 d-fragments, K-dim = 32 keys ---------------
            #pragma unroll
            for (int f = 0; f < 4; ++f) {
                // B fragment: col = d = 16f + lane%16, k = 16*hi + e
                // contiguous 16 bf16 row of transposed V in LDS
                const v16bf* vb = (const v16bf*)
                    &VTl[(f * 16 + ln16) * KTILE + hi * 16];
                acc[f] = __builtin_amdgcn_wmma_f32_16x16x32_bf16(
                            false, pa, false, *vb, (short)0, acc[f], false, false);
            }
        }
    }

    // ---- epilogue: O / l, write fp32 (coalesced across 16-lane groups) ----
    float inv[8];
    #pragma unroll
    for (int r = 0; r < 8; ++r) inv[r] = 1.0f / l[r];
    #pragma unroll
    for (int f = 0; f < 4; ++f) {
        #pragma unroll
        for (int r = 0; r < 8; ++r) {
            const int q = q0 + r + 8 * hi;
            Op[(size_t)q * D_DIM + f * 16 + ln16] = acc[f][r] * inv[r];
        }
    }
}

extern "C" void kernel_launch(void* const* d_in, const int* in_sizes, int n_in,
                              void* d_out, int out_size, void* d_ws, size_t ws_size,
                              hipStream_t stream)
{
    const float* Q = (const float*)d_in[0];
    const float* K = (const float*)d_in[1];
    const float* V = (const float*)d_in[2];
    // d_in[3] is the causal bool mask; handled analytically in the kernel.
    float* O = (float*)d_out;

    const int BH      = 2 * 16;
    const int qblocks = S_LEN / BLOCKQ;          // 32
    dim3 grid(BH * qblocks);                     // 1024 workgroups
    dim3 block(128);                             // 4 waves (wave32)
    fa_fwd_kernel<<<grid, block, 0, stream>>>(Q, K, V, O);
}
